// Unified_Cross_Entropy_Loss_10849087389705
// MI455X (gfx1250) — matrix operand order repaired and checked by reading, not compiled, not run on hardware
//
#include <hip/hip_runtime.h>
#include <hip/hip_bf16.h>
#include <math.h>

// Problem constants (match reference)
#define B_ 512
#define D_ 512
#define C_ 100000
#define S_SCALE 64.0f
#define M_MARGIN 0.4f
#define EPS_ 1e-5f

#define BT_ 64    // batch rows per block
#define CT_ 128   // class cols per block
#define KT_ 32    // K per WMMA step
#define NK_ (D_ / KT_)

typedef __attribute__((ext_vector_type(16))) _Float16 v16h;
typedef __attribute__((ext_vector_type(8)))  _Float16 v8h;
typedef __attribute__((ext_vector_type(4)))  _Float16 v4h;
typedef __attribute__((ext_vector_type(8)))  float    v8f;

__device__ __forceinline__ float softplus_f(float x) {
  // log1p(exp(x)), numerically stable over the clamped range [-64, 64]
  return fmaxf(x, 0.0f) + log1pf(__expf(-fabsf(x)));
}

__global__ void uce_zero_out(float* o) {
  if (threadIdx.x == 0) o[0] = 0.0f;
}

// Per-iteration global staging registers (kept small so the K loop stays
// register-bounded and occupancy stays high on the 23.3 TB/s HBM stream).
struct Stage {
  float4 a[2];   // 2 input rows  x float4
  float4 w[4];   // 4 weight rows x float4
};

__device__ __forceinline__ void stage_load(Stage& s,
                                           const float* __restrict__ input,
                                           const float* __restrict__ weight,
                                           int by, int bx, int ar, int ac, int k0) {
  #pragma unroll
  for (int j = 0; j < 2; ++j)
    s.a[j] = *(const float4*)(input + (size_t)(by + ar + 32 * j) * D_ + k0 + ac);
  #pragma unroll
  for (int j = 0; j < 4; ++j) {
    int gr = bx + ar + 32 * j;
    gr = gr < C_ ? gr : C_ - 1;            // branchless tail: clamp, mask later
    const float* wp = weight + (size_t)gr * D_ + k0 + ac;
    s.w[j] = *(const float4*)wp;
    // Branchless L2 prefetch of the next K tile. Rows are contiguous, so
    // wp+KT_ stays inside the buffer except <=128B past its very end, where
    // a speculative global_prefetch_b8 is architecturally dropped.
    __builtin_prefetch(wp + KT_, 0, 1);
  }
}

__device__ __forceinline__ void stage_store(const Stage& s,
                                            _Float16 (* __restrict__ As)[KT_],
                                            _Float16 (* __restrict__ Ws)[KT_],
                                            int ar, int ac,
                                            float* issq, float* wssq) {
  #pragma unroll
  for (int j = 0; j < 2; ++j) {
    const float4 v = s.a[j];
    _Float16 h[4] = {(_Float16)v.x, (_Float16)v.y, (_Float16)v.z, (_Float16)v.w};
    *(v4h*)&As[ar + 32 * j][ac] = *(v4h*)h;
    issq[j] += v.x * v.x + v.y * v.y + v.z * v.z + v.w * v.w;
  }
  #pragma unroll
  for (int j = 0; j < 4; ++j) {
    const float4 v = s.w[j];
    _Float16 h[4] = {(_Float16)v.x, (_Float16)v.y, (_Float16)v.z, (_Float16)v.w};
    *(v4h*)&Ws[ar + 32 * j][ac] = *(v4h*)h;
    wssq[j] += v.x * v.x + v.y * v.y + v.z * v.z + v.w * v.w;
  }
}

__launch_bounds__(256)
__global__ void uce_loss_kernel(const float* __restrict__ input,   // [B, D]
                                const int*   __restrict__ label,   // [B]
                                const float* __restrict__ weight,  // [C, D]
                                const float* __restrict__ bias,    // [1]
                                float* __restrict__ out)           // [1]
{
  // Double-buffered staging tiles: 8 KB + 16 KB
  __shared__ __align__(16) _Float16 As[2][BT_][KT_];
  __shared__ __align__(16) _Float16 Ws[2][CT_][KT_];
  __shared__ float s_iscale[BT_];
  __shared__ float s_wscale[CT_];
  __shared__ int   s_lab[BT_];
  __shared__ float s_red[8];

  const int tid   = threadIdx.x;
  const int wave  = tid >> 5;
  const int lane  = tid & 31;
  const int waveM = wave >> 2;          // 0..1  (batch dir)
  const int waveN = wave & 3;           // 0..3  (class dir)
  const int by    = blockIdx.y * BT_;
  const int bx    = blockIdx.x * CT_;

  if (tid < BT_) s_lab[tid] = label[by + tid];

  // Loader mapping: 8 lanes per row (float4 each), lane-aligned groups of 8.
  const int ar = tid >> 3;              // 0..31
  const int ac = (tid & 7) * 4;         // 0,4,...,28

  float issq[2] = {0.0f, 0.0f};
  float wssq[4] = {0.0f, 0.0f, 0.0f, 0.0f};

  v8f acc[2][2] = {};

  // Fragment addressing (ISA VGPR layouts for v_wmma_f32_16x16x32_f16):
  //  A 16x32 f16: lane<16 holds K 0..7 / 16..23 of row (lane&15); lane>=16 K+8
  //  B 32x16 f16: lane<16 holds K 0..7 / 16..23 of col (lane&15); lane>=16 K+8
  const int mrow  = lane & 15;
  const int khalf = (lane >> 4) * 8;
  const int kb    = (lane >> 4) * 16;

  union V16 { v16h v; v8h h[2]; };

  // One 2x2 WMMA step from LDS buffer `buf`
  auto wmma_step = [&](int buf) {
    V16 af[2], bf[2];
    #pragma unroll
    for (int tm = 0; tm < 2; ++tm) {
      const int r = waveM * 32 + tm * 16 + mrow;
      af[tm].h[0] = *(const v8h*)&As[buf][r][khalf];
      af[tm].h[1] = *(const v8h*)&As[buf][r][16 + khalf];
    }
    #pragma unroll
    for (int tn = 0; tn < 2; ++tn) {
      const int r = waveN * 32 + tn * 16 + mrow;
      bf[tn].h[0] = *(const v8h*)&Ws[buf][r][khalf];
      bf[tn].h[1] = *(const v8h*)&Ws[buf][r][16 + khalf];
    }
    #pragma unroll
    for (int tm = 0; tm < 2; ++tm)
      #pragma unroll
      for (int tn = 0; tn < 2; ++tn)
        acc[tm][tn] = __builtin_amdgcn_wmma_f32_16x16x32_f16(
            false, af[tm].v, false, bf[tn].v,
            (short)0, acc[tm][tn], false, false);
  };

  // ---- pipeline prologue: stage K-tile 0 into buffer 0 ----
  Stage st;
  stage_load(st, input, weight, by, bx, ar, ac, 0);
  stage_store(st, As[0], Ws[0], ar, ac, issq, wssq);
  __syncthreads();

  // ---- steady state: always load/convert tile kt+1 around the WMMAs ----
  #pragma unroll 2
  for (int kt = 0; kt < NK_ - 1; ++kt) {
    const int cur = kt & 1;
    // Issue next tile's global loads before the WMMAs (overlap HBM w/ matrix)
    stage_load(st, input, weight, by, bx, ar, ac, (kt + 1) * KT_);
    wmma_step(cur);
    stage_store(st, As[cur ^ 1], Ws[cur ^ 1], ar, ac, issq, wssq);
    __syncthreads();
  }
  // ---- pipeline epilogue: last K tile, no staging ----
  wmma_step((NK_ - 1) & 1);

  // ---- reduce sum-of-squares across the 8 loader lanes per row ----
  #pragma unroll
  for (int off = 4; off > 0; off >>= 1) {
    #pragma unroll
    for (int j = 0; j < 2; ++j) issq[j] += __shfl_xor(issq[j], off, 32);
    #pragma unroll
    for (int j = 0; j < 4; ++j) wssq[j] += __shfl_xor(wssq[j], off, 32);
  }
  if ((tid & 7) == 0) {
    #pragma unroll
    for (int j = 0; j < 2; ++j)
      s_iscale[ar + 32 * j] = 1.0f / fmaxf(sqrtf(issq[j]), EPS_);
    #pragma unroll
    for (int j = 0; j < 4; ++j)
      s_wscale[ar + 32 * j] = 1.0f / fmaxf(sqrtf(wssq[j]), EPS_);
  }
  __syncthreads();

  // ---- fused loss epilogue: cos -> softplus losses -> scalar reduce ----
  const float bia  = bias[0];
  const int   nloc = lane & 15;
  const int   rofs = 8 * (lane >> 4);   // C/D layout: row = k + 8*(lane>=16)
  float lsum = 0.0f;

  #pragma unroll
  for (int tm = 0; tm < 2; ++tm) {
    #pragma unroll
    for (int tn = 0; tn < 2; ++tn) {
      const int cl = waveN * 32 + tn * 16 + nloc;
      const int gc = bx + cl;
      if (gc >= C_) continue;                 // mask clamped tail columns
      const float wsc = s_wscale[cl];
      #pragma unroll
      for (int k = 0; k < 8; ++k) {
        const int rl = waveM * 32 + tm * 16 + k + rofs;
        const float cosv = acc[tm][tn][k] * s_iscale[rl] * wsc;
        const float zn = fminf(fmaxf(S_SCALE * cosv - bia, -S_SCALE), S_SCALE);
        const float nl = softplus_f(zn);      // log1p(exp(zn))
        lsum += nl;
        if (s_lab[rl] == gc) {
          const float zp = fminf(fmaxf(S_SCALE * (cosv - M_MARGIN) - bia,
                                       -S_SCALE), S_SCALE);
          lsum += softplus_f(-zp) - nl;       // swap in p_loss for the label
        }
      }
    }
  }

  // wave reduce, then block reduce, then one atomic per block
  #pragma unroll
  for (int off = 16; off > 0; off >>= 1) lsum += __shfl_down(lsum, off, 32);
  if (lane == 0) s_red[wave] = lsum;
  __syncthreads();
  if (tid == 0) {
    float t = 0.0f;
    #pragma unroll
    for (int w = 0; w < 8; ++w) t += s_red[w];
    atomicAdd(out, t * (1.0f / (float)B_));   // mean over batch
  }
}

extern "C" void kernel_launch(void* const* d_in, const int* in_sizes, int n_in,
                              void* d_out, int out_size, void* d_ws, size_t ws_size,
                              hipStream_t stream) {
  (void)in_sizes; (void)n_in; (void)out_size; (void)d_ws; (void)ws_size;
  const float* input  = (const float*)d_in[0];
  const int*   label  = (const int*)d_in[1];
  const float* weight = (const float*)d_in[2];
  const float* bias   = (const float*)d_in[3];
  float* out = (float*)d_out;

  uce_zero_out<<<1, 32, 0, stream>>>(out);

  dim3 grid((C_ + CT_ - 1) / CT_, B_ / BT_);   // 782 x 8 blocks
  uce_loss_kernel<<<grid, 256, 0, stream>>>(input, label, weight, bias, out);
}
